// SimpleRNN_12824772346006
// MI455X (gfx1250) — compile-verified, matrix operand order checked by reading, not run on hardware
//
#include <hip/hip_runtime.h>
#include <hip/hip_bf16.h>

// ---------------------------------------------------------------------------
// Stacked tanh-RNN (B=64, T=256, I=128, H=1024, L=3, O=2) for gfx1250.
// Persistent kernel, software grid barrier, bf16 WMMA 16x16x32 with f32 acc.
// Each wave computes a 16x64 output slab (1 M-tile x 4 N-tiles): A fragment
// reused across 4 WMMAs, 4 independent accumulator chains for ILP.
// K-loops kept as real loops (#pragma unroll 1) so weight fragments are NOT
// hoisted across the time loop (previous round spilled ~300 VGPRs to scratch).
// ---------------------------------------------------------------------------

typedef __attribute__((ext_vector_type(16))) __bf16 v16bf;
typedef __attribute__((ext_vector_type(8)))  float  v8f;

constexpr int BB = 64;    // batch
constexpr int TT = 256;   // time steps
constexpr int II = 128;   // input size
constexpr int HH = 1024;  // hidden size
constexpr int LL = 3;     // layers
constexpr int OO = 2;     // output size

constexpr int NBLOCKS  = 8;    // persistent blocks (all co-resident, 8 arrivals/sync)
constexpr int NTHREADS = 256;  // 8 wave32 per block -> 64 waves total
// 64 waves x (16 rows x 64 cols) = full [64,1024] state each layer.

#define DEVFN __device__ __forceinline__

// ---- WMMA helpers ----------------------------------------------------------

DEVFN v8f wmma_bf16(v16bf a, v16bf b, v8f c) {
    // emits v_wmma_f32_16x16x32_bf16
    return __builtin_amdgcn_wmma_f32_16x16x32_bf16(
        /*neg_a=*/false, a, /*neg_b=*/false, b,
        /*c_mod=*/(short)0, c, /*reuse_a=*/false, /*reuse_b=*/false);
}

// Load a 16x32 bf16 fragment (A layout; B layout is its mirror, so the same
// loader serves row-major W[H,K] when computing A*W^T).
// CDNA5 ISA 7.12.2: lanes 0-15 hold row m=lane, K=[k0..k0+7, k0+16..k0+23];
// lanes 16-31 same rows, K shifted +8. Two global_load_b128 per lane.
DEVFN v16bf load_frag(const __hip_bfloat16* __restrict__ p, int ld, int row, int k0) {
    const int lane = threadIdx.x & 31;
    const int r = row + (lane & 15);
    const int c = k0 + ((lane & 16) ? 8 : 0);
    const __hip_bfloat16* base = p + (size_t)r * ld + c;
    union { uint4 q[2]; v16bf v; } u;
    u.q[0] = *reinterpret_cast<const uint4*>(base);        // K = c .. c+7
    u.q[1] = *reinterpret_cast<const uint4*>(base + 16);   // K = c+16 .. c+23
    return u.v;
}

// ---- 4-accumulator slab ----------------------------------------------------

struct Acc4 { v8f a0, a1, a2, a3; };

DEVFN void mma4(Acc4& acc,
                const __hip_bfloat16* __restrict__ A, int lda, int mtile,
                const __hip_bfloat16* __restrict__ W, int ldw, int ncol0,
                int k0) {
    v16bf af = load_frag(A, lda, mtile, k0);
    v16bf b0 = load_frag(W, ldw, ncol0 +  0, k0);
    v16bf b1 = load_frag(W, ldw, ncol0 + 16, k0);
    v16bf b2 = load_frag(W, ldw, ncol0 + 32, k0);
    v16bf b3 = load_frag(W, ldw, ncol0 + 48, k0);
    acc.a0 = wmma_bf16(af, b0, acc.a0);
    acc.a1 = wmma_bf16(af, b1, acc.a1);
    acc.a2 = wmma_bf16(af, b2, acc.a2);
    acc.a3 = wmma_bf16(af, b3, acc.a3);
}

// ---- software grid barrier (agent scope) -----------------------------------

DEVFN void grid_sync(unsigned* cnt, unsigned* gen) {
    __syncthreads();
    if (threadIdx.x == 0) {
        __threadfence();  // make this block's global writes visible device-wide
        unsigned g = __hip_atomic_load(gen, __ATOMIC_RELAXED, __HIP_MEMORY_SCOPE_AGENT);
        unsigned arrived = __hip_atomic_fetch_add(cnt, 1u, __ATOMIC_ACQ_REL,
                                                  __HIP_MEMORY_SCOPE_AGENT);
        if (arrived == (unsigned)(NBLOCKS - 1)) {
            __hip_atomic_store(cnt, 0u, __ATOMIC_RELAXED, __HIP_MEMORY_SCOPE_AGENT);
            __hip_atomic_fetch_add(gen, 1u, __ATOMIC_ACQ_REL, __HIP_MEMORY_SCOPE_AGENT);
        } else {
            while (__hip_atomic_load(gen, __ATOMIC_ACQUIRE,
                                     __HIP_MEMORY_SCOPE_AGENT) == g) {
                __builtin_amdgcn_s_sleep(1);
            }
        }
    }
    __syncthreads();
}

// ---- epilogue: bias + tanh + bf16 store (optional f32 copy for final FC) ---
// C/D layout: VGPR i -> M = mtile + i (+8 for lanes 16-31), N = ntile + (lane&15)

DEVFN void store_one(v8f acc, const float* __restrict__ biasA,
                     const float* __restrict__ biasB,
                     __hip_bfloat16* __restrict__ hdst,
                     float* __restrict__ fdst, int mtile, int ntile) {
    const int lane  = threadIdx.x & 31;
    const int n     = ntile + (lane & 15);
    const float bias = biasA[n] + biasB[n];
    const int mbase = mtile + ((lane & 16) ? 8 : 0);
#pragma unroll
    for (int i = 0; i < 8; ++i) {
        float v = tanhf(acc[i] + bias);
        hdst[(size_t)(mbase + i) * HH + n] = __float2bfloat16(v);
        if (fdst) fdst[(size_t)(mbase + i) * HH + n] = v;
    }
}

DEVFN void store_slab(Acc4& acc, const float* __restrict__ biasA,
                      const float* __restrict__ biasB,
                      __hip_bfloat16* __restrict__ hdst,
                      float* __restrict__ fdst, int mtile, int ncol0) {
    store_one(acc.a0, biasA, biasB, hdst, fdst, mtile, ncol0 +  0);
    store_one(acc.a1, biasA, biasB, hdst, fdst, mtile, ncol0 + 16);
    store_one(acc.a2, biasA, biasB, hdst, fdst, mtile, ncol0 + 32);
    store_one(acc.a3, biasA, biasB, hdst, fdst, mtile, ncol0 + 48);
}

// ---- f32 -> bf16 conversion ------------------------------------------------

__global__ void f32_to_bf16_kernel(const float* __restrict__ s,
                                   __hip_bfloat16* __restrict__ d, int n) {
    int i = blockIdx.x * blockDim.x + threadIdx.x;
    const int stride = gridDim.x * blockDim.x;
    for (; i < n; i += stride) d[i] = __float2bfloat16(s[i]);
}

// ---- persistent RNN kernel -------------------------------------------------

__global__ void __launch_bounds__(NTHREADS)
rnn_persistent(const __hip_bfloat16* __restrict__ xbf,   // [B,T,I]
               const __hip_bfloat16* __restrict__ wx0,   // [H,I]
               const __hip_bfloat16* __restrict__ wh0,   // [H,H]
               const __hip_bfloat16* __restrict__ wxs,   // [L-1,H,H]
               const __hip_bfloat16* __restrict__ whs,   // [L-1,H,H]
               const float* __restrict__ bx0, const float* __restrict__ bh0,
               const float* __restrict__ bxs, const float* __restrict__ bhs,
               const float* __restrict__ fcw, const float* __restrict__ fcb,
               __hip_bfloat16* __restrict__ hbuf,        // [2,L,B,H] ping-pong
               float* __restrict__ htop,                 // [B,H] top layer f32
               float* __restrict__ out,                  // [B,O]
               unsigned* bar_cnt, unsigned* bar_gen) {
    const int wave  = (blockIdx.x * NTHREADS + threadIdx.x) >> 5;
    const int mtile = (wave >> 4) << 4;  // 0,16,32,48   (batch rows)
    const int ncol0 = (wave & 15) << 6;  // 0..960       (64 hidden cols per wave)

    int cur = 0;
    for (int t = 0; t < TT; ++t) {
        const int nxt = cur ^ 1;

        // ---------------- layer 0: tanh(x_t wx0^T + h0 wh0^T + b) ----------
        {
            Acc4 acc = {};
            const __hip_bfloat16* xrow = xbf + (size_t)t * II;  // row stride T*I
#pragma unroll 1
            for (int k0 = 0; k0 < II; k0 += 32)
                mma4(acc, xrow, TT * II, mtile, wx0, II, ncol0, k0);
            const __hip_bfloat16* h_old = hbuf + ((size_t)cur * LL + 0) * BB * HH;
#pragma unroll 1
            for (int k0 = 0; k0 < HH; k0 += 32)
                mma4(acc, h_old, HH, mtile, wh0, HH, ncol0, k0);
            store_slab(acc, bx0, bh0,
                       hbuf + ((size_t)nxt * LL + 0) * BB * HH, nullptr,
                       mtile, ncol0);
        }
        grid_sync(bar_cnt, bar_gen);

        // ---------------- layers 1..L-1 ------------------------------------
#pragma unroll 1
        for (int l = 0; l < LL - 1; ++l) {
            Acc4 acc = {};
            const __hip_bfloat16* a_prev = hbuf + ((size_t)nxt * LL + l) * BB * HH;     // new h[l]
            const __hip_bfloat16* a_self = hbuf + ((size_t)cur * LL + l + 1) * BB * HH; // old h[l+1]
            const __hip_bfloat16* wxl = wxs + (size_t)l * HH * HH;
            const __hip_bfloat16* whl = whs + (size_t)l * HH * HH;
#pragma unroll 1
            for (int k0 = 0; k0 < HH; k0 += 32)
                mma4(acc, a_prev, HH, mtile, wxl, HH, ncol0, k0);
#pragma unroll 1
            for (int k0 = 0; k0 < HH; k0 += 32)
                mma4(acc, a_self, HH, mtile, whl, HH, ncol0, k0);
            store_slab(acc, bxs + l * HH, bhs + l * HH,
                       hbuf + ((size_t)nxt * LL + l + 1) * BB * HH,
                       (l == LL - 2) ? htop : nullptr, mtile, ncol0);
            grid_sync(bar_cnt, bar_gen);
        }
        cur = nxt;
    }

    // ---------------- final FC: out = htop @ fcw^T + fcb (tiny) -------------
    if (blockIdx.x == 0) {
        for (int idx = threadIdx.x; idx < BB * OO; idx += NTHREADS) {
            const int b = idx / OO, o = idx % OO;
            float s = fcb[o];
            const float* hp = htop + (size_t)b * HH;
            const float* wp = fcw + (size_t)o * HH;
            for (int k = 0; k < HH; ++k) s = fmaf(hp[k], wp[k], s);
            out[idx] = s;
        }
    }
}

// ---------------------------------------------------------------------------

extern "C" void kernel_launch(void* const* d_in, const int* in_sizes, int n_in,
                              void* d_out, int out_size, void* d_ws, size_t ws_size,
                              hipStream_t stream) {
    (void)in_sizes; (void)n_in; (void)out_size; (void)ws_size;

    const float* x   = (const float*)d_in[0];   // [B,T,I]
    const float* wx0 = (const float*)d_in[1];   // [H,I]
    const float* bx0 = (const float*)d_in[2];   // [H]
    const float* wh0 = (const float*)d_in[3];   // [H,H]
    const float* bh0 = (const float*)d_in[4];   // [H]
    const float* wx  = (const float*)d_in[5];   // [L-1,H,H]
    const float* bx  = (const float*)d_in[6];   // [L-1,H]
    const float* wh  = (const float*)d_in[7];   // [L-1,H,H]
    const float* bh  = (const float*)d_in[8];   // [L-1,H]
    const float* fcw = (const float*)d_in[9];   // [O,H]
    const float* fcb = (const float*)d_in[10];  // [O]
    float* out = (float*)d_out;

    // --- workspace carve-out (256B aligned chunks) ---
    char* ws = (char*)d_ws;
    size_t off = 0;
    auto carve = [&](size_t bytes) -> void* {
        void* p = ws + off;
        off += (bytes + 255) & ~(size_t)255;
        return p;
    };
    unsigned* bar = (unsigned*)carve(256);                        // cnt @ +0, gen @ +64
    __hip_bfloat16* xbf   = (__hip_bfloat16*)carve((size_t)BB * TT * II * 2);
    __hip_bfloat16* wx0bf = (__hip_bfloat16*)carve((size_t)HH * II * 2);
    __hip_bfloat16* wh0bf = (__hip_bfloat16*)carve((size_t)HH * HH * 2);
    __hip_bfloat16* wxbf  = (__hip_bfloat16*)carve((size_t)(LL - 1) * HH * HH * 2);
    __hip_bfloat16* whbf  = (__hip_bfloat16*)carve((size_t)(LL - 1) * HH * HH * 2);
    __hip_bfloat16* hbuf  = (__hip_bfloat16*)carve((size_t)2 * LL * BB * HH * 2);
    float*          htop  = (float*)carve((size_t)BB * HH * 4);

    unsigned* bar_cnt = bar;
    unsigned* bar_gen = bar + 16;  // separate cacheline

    // zero barrier state and initial hidden state (h0 = 0), every call
    hipMemsetAsync(bar, 0, 256, stream);
    hipMemsetAsync(hbuf, 0, (size_t)2 * LL * BB * HH * 2, stream);

    // f32 -> bf16 conversions (weights fit in L2; done fresh each call)
    auto cvt = [&](const float* s, __hip_bfloat16* d, int n) {
        int blocks = (n + 255) / 256;
        if (blocks > 2048) blocks = 2048;
        f32_to_bf16_kernel<<<blocks, 256, 0, stream>>>(s, d, n);
    };
    cvt(x,   xbf,   BB * TT * II);
    cvt(wx0, wx0bf, HH * II);
    cvt(wh0, wh0bf, HH * HH);
    cvt(wx,  wxbf,  (LL - 1) * HH * HH);
    cvt(wh,  whbf,  (LL - 1) * HH * HH);

    rnn_persistent<<<NBLOCKS, NTHREADS, 0, stream>>>(
        xbf, wx0bf, wh0bf, wxbf, whbf,
        bx0, bh0, bx, bh, fcw, fcb,
        hbuf, htop, out, bar_cnt, bar_gen);
}
